// SwinTransformerBlock_45260365365696
// MI455X (gfx1250) — compile-verified, matrix operand order checked by reading, not run on hardware
//
#include <hip/hip_runtime.h>
#include <hip/hip_bf16.h>

// ---------------------------------------------------------------------------
// Swin block for gfx1250 (MI455X): f16 WMMA (v_wmma_f32_16x16x32_f16) GEMMs,
// TDM (tensor_load_to_lds) staging of all LDS panels, wave32 blocks,
// LDS-tiled 64x64 output tiles, fused windowing/shift/residual epilogues.
// ---------------------------------------------------------------------------

typedef __attribute__((ext_vector_type(16))) _Float16 v16h;
typedef __attribute__((ext_vector_type(8)))  _Float16 v8h;
typedef __attribute__((ext_vector_type(8)))  float    v8f;
typedef __attribute__((ext_vector_type(4)))  unsigned int u32x4;
typedef __attribute__((ext_vector_type(8)))  int          i32x8;
typedef __attribute__((ext_vector_type(4)))  int          i32x4;

constexpr int   kB      = 4;
constexpr int   kHW     = 224;
constexpr int   kT      = kHW * kHW;          // 50176 tokens per image
constexpr int   kM      = kB * kT;            // 200704 token rows
constexpr int   kC      = 192;
constexpr int   kNH     = 6;
constexpr int   kDH     = 32;
constexpr int   kWS     = 8;
constexpr int   kN      = 64;                 // tokens per window
constexpr int   kShift  = 4;
constexpr int   kNWx    = 28;                 // windows per image edge
constexpr int   kNWimg  = kNWx * kNWx;        // 784
constexpr int   kNW     = kB * kNWimg;        // 3136
constexpr int   kQKN    = 3 * kC;             // 576
constexpr int   kHID    = 384;
constexpr float kScale  = 0.17677669529663687f; // 32^-0.5

// ---------------------------------------------------------------------------
// Tensor Data Mover: DMA a 2D tile of f16 (tile_w x tile_h elements, row pitch
// `stride` elements) from global memory into LDS. Builds the D# per ISA
// CDNA5 ISA and issues TENSOR_LOAD_TO_LDS (tracked by TENSORcnt).
// This toolchain's builtin takes 6 args (extra zero-filled int32x8 before the
// cache-policy word). Call from a single wave per block; follow with
// s_wait_tensorcnt + barrier.
// ---------------------------------------------------------------------------
__device__ inline void tdm_load_2d_f16(void* lds_dst, const _Float16* src,
                                       int tile_w, int tile_h, int stride) {
  const unsigned long long ga = (unsigned long long)(uintptr_t)src;
  const unsigned int lds = (unsigned int)(uintptr_t)lds_dst;   // LDS byte addr

  u32x4 g0;
  g0[0] = 1u;                                  // count=1, user descriptor
  g0[1] = lds;                                 // lds_addr[31:0]
  g0[2] = (unsigned int)ga;                    // global_addr[31:0]
  g0[3] = (unsigned int)((ga >> 32) & 0x01FFFFFFu) | (2u << 30); // [56:32]+type=2

  i32x8 g1;
  g1[0] = 1 << 16;                             // data_size=1 (2 bytes)
  g1[1] = (tile_w & 0xFFFF) << 16;             // tensor_dim0[15:0]
  g1[2] = (tile_h & 0xFFFF) << 16;             // dim0 hi=0 | tensor_dim1[15:0]
  g1[3] = (tile_w & 0xFFFF) << 16;             // dim1 hi=0 | tile_dim0
  g1[4] = (tile_h & 0xFFFF);                   // tile_dim1 | tile_dim2=0
  g1[5] = stride;                              // tensor_dim0_stride[31:0]
  g1[6] = 0;                                   // stride hi | dim1_stride lo
  g1[7] = 0;                                   // dim1_stride hi

  i32x4 g2 = {1, 0, 0, 0};                     // tensor_dim2=1 (unused dims 0)
  i32x4 g3 = {0, 0, 0, 0};
  i32x8 g4 = {0, 0, 0, 0, 0, 0, 0, 0};         // unused trailing group (zeroed)
  __builtin_amdgcn_tensor_load_to_lds(g0, g1, g2, g3, g4, 0);
}

// ---------------------------------------------------------------------------
// WMMA helpers: 16x16x32 f16 -> f32, fragments loaded as two 16B LDS reads
// matching the documented 16-bit A/B VGPR striping (lane&15 = row,
// lane>>4 selects the K-half; VGPR0..3 = K 0..7(+kh), VGPR4..7 = K 16..23(+kh)).
// ---------------------------------------------------------------------------
__device__ inline v8f wmma_f16(v16h a, v16h b, v8f c) {
  return __builtin_amdgcn_wmma_f32_16x16x32_f16(false, a, false, b,
                                                (short)0, c, false, false);
}

__device__ inline v16h load_frag(const _Float16* p, int ld, int lane) {
  const int r  = lane & 15;
  const int kh = (lane >> 4) << 3;
  const _Float16* q = p + r * ld + kh;
  union { v16h v; v8h h[2]; } u;
  u.h[0] = *(const v8h*)(q);
  u.h[1] = *(const v8h*)(q + 16);
  return u.v;
}

__device__ inline float wave_sum(float v) {
#pragma unroll
  for (int m = 16; m > 0; m >>= 1) v += __shfl_xor(v, m, 32);
  return v;
}

// ---------------------------------------------------------------------------
// LN1 + cyclic shift + window partition: one wave32 per token.
// Output: xw[windowRow R][c] in f16, R enumerates (b, wy, wx, i, j).
// ---------------------------------------------------------------------------
__global__ __launch_bounds__(256) void k_ln1_window(
    const float* __restrict__ x, const float* __restrict__ g,
    const float* __restrict__ b, _Float16* __restrict__ xw) {
  const int gid  = blockIdx.x * 256 + threadIdx.x;
  const int R    = gid >> 5;
  const int lane = gid & 31;
  const int t    = R & 63;
  const int win  = R >> 6;
  const int wx   = win % kNWx;
  const int wy   = (win / kNWx) % kNWx;
  const int bb   = win / kNWimg;
  const int sh   = (wy * kWS + (t >> 3) + kShift) % kHW;   // roll by -shift
  const int sw   = (wx * kWS + (t & 7)  + kShift) % kHW;
  const float* row = x + ((size_t)bb * kT + (size_t)sh * kHW + sw) * kC;

  float v[6]; float s = 0.f;
#pragma unroll
  for (int i = 0; i < 6; i++) { v[i] = row[lane + 32 * i]; s += v[i]; }
  const float mu = wave_sum(s) * (1.f / kC);
  float q = 0.f;
#pragma unroll
  for (int i = 0; i < 6; i++) { float d = v[i] - mu; q += d * d; }
  const float rstd = rsqrtf(wave_sum(q) * (1.f / kC) + 1e-5f);

  _Float16* out = xw + (size_t)R * kC;
#pragma unroll
  for (int i = 0; i < 6; i++) {
    const int c = lane + 32 * i;
    out[c] = (_Float16)((v[i] - mu) * rstd * g[c] + b[c]);
  }
}

// LN2: token order, f32 in (y = x + attn branch), f16 out.
__global__ __launch_bounds__(256) void k_ln2(
    const float* __restrict__ y, const float* __restrict__ g,
    const float* __restrict__ b, _Float16* __restrict__ h16) {
  const int gid  = blockIdx.x * 256 + threadIdx.x;
  const int R    = gid >> 5;
  const int lane = gid & 31;
  const float* row = y + (size_t)R * kC;

  float v[6]; float s = 0.f;
#pragma unroll
  for (int i = 0; i < 6; i++) { v[i] = row[lane + 32 * i]; s += v[i]; }
  const float mu = wave_sum(s) * (1.f / kC);
  float q = 0.f;
#pragma unroll
  for (int i = 0; i < 6; i++) { float d = v[i] - mu; q += d * d; }
  const float rstd = rsqrtf(wave_sum(q) * (1.f / kC) + 1e-5f);

  _Float16* out = h16 + (size_t)R * kC;
#pragma unroll
  for (int i = 0; i < 6; i++) {
    const int c = lane + 32 * i;
    out[c] = (_Float16)((v[i] - mu) * rstd * g[c] + b[c]);
  }
}

// Weight prep: src[K][N] f32 -> dst[N][K] f16 (transposed for contiguous B frags)
__global__ void k_wprep(const float* __restrict__ src, _Float16* __restrict__ dst,
                        int K, int N) {
  const int i = blockIdx.x * 256 + threadIdx.x;
  if (i >= K * N) return;
  const int k = i / N, n = i % N;
  dst[(size_t)n * K + k] = (_Float16)src[i];
}

// Relative position bias table -> dense [head][64][64]
__global__ void k_biasprep(const float* __restrict__ table,
                           float* __restrict__ biasTab) {
  const int i = blockIdx.x * 256 + threadIdx.x;    // < 6*4096, exact grid
  const int h = i >> 12, r = i & 4095;
  const int n = r >> 6, m = r & 63;                // n = query, m = key
  const int di = (n >> 3) - (m >> 3) + (kWS - 1);
  const int dj = (n & 7)  - (m & 7)  + (kWS - 1);
  const int idx = di * (2 * kWS - 1) + dj;
  biasTab[i] = table[idx * kNH + h];
}

// ---------------------------------------------------------------------------
// Generic WMMA GEMM: C[M,N] = A16[M,K] @ BT16[N,K]^T, 64x64 tile per block.
// 8 waves: wave = (nj<<2)|mi -> (mi, {nj*2, nj*2+1}) 16x16 subtiles.
// K-panels of 192 DMA'd into LDS by the TDM (wave 0 issues, TENSORcnt-waited,
// then barrier). Epilogue functor handles bias/fusion.
// ---------------------------------------------------------------------------
template<int K, typename Epi>
__global__ __launch_bounds__(256) void k_gemm64(
    const _Float16* __restrict__ A, const _Float16* __restrict__ BT, Epi epi) {
  constexpr int BK = 192;
  static_assert(K % BK == 0, "K must be a multiple of the panel size");
  __shared__ __align__(16) _Float16 sA[64 * BK];
  __shared__ __align__(16) _Float16 sB[64 * BK];

  const int m0   = blockIdx.x * 64;
  const int n0   = blockIdx.y * 64;
  const int wave = threadIdx.x >> 5;
  const int lane = threadIdx.x & 31;
  const int mi   = wave & 3;
  const int nj   = (wave >> 2) << 1;

  v8f c0 = {}, c1 = {};

  for (int k0 = 0; k0 < K; k0 += BK) {
    if (wave == 0) {                       // one TDM issue per panel per block
      tdm_load_2d_f16(sA, A  + (size_t)m0 * K + k0, BK, 64, K);
      tdm_load_2d_f16(sB, BT + (size_t)n0 * K + k0, BK, 64, K);
      __builtin_amdgcn_s_wait_tensorcnt(0);
    }
    __syncthreads();

    const _Float16* aB  = sA + mi * 16 * BK;
    const _Float16* b0B = sB + (nj + 0) * 16 * BK;
    const _Float16* b1B = sB + (nj + 1) * 16 * BK;
#pragma unroll
    for (int kk = 0; kk < BK; kk += 32) {
      v16h a  = load_frag(aB  + kk, BK, lane);
      v16h b0 = load_frag(b0B + kk, BK, lane);
      v16h b1 = load_frag(b1B + kk, BK, lane);
      c0 = wmma_f16(a, b0, c0);
      c1 = wmma_f16(a, b1, c1);
    }
    __syncthreads();
  }

  // D layout: VGPR r holds row (r + 8*(lane>>4)), col = lane&15.
  const int rbase = m0 + mi * 16 + ((lane >> 4) << 3);
  const int ca    = n0 + nj * 16 + (lane & 15);
#pragma unroll
  for (int r = 0; r < 8; r++) {
    epi(rbase + r, ca,      c0[r]);
    epi(rbase + r, ca + 16, c1[r]);
  }
}

// Epilogues ------------------------------------------------------------------
struct EpiQKV {                 // + qkv bias, f16 out [M][576]
  const float* bias; _Float16* out;
  __device__ void operator()(int row, int col, float v) const {
    out[(size_t)row * kQKN + col] = (_Float16)(v + bias[col]);
  }
};
struct EpiProj {                // + proj bias, un-window/un-shift, + residual
  const float* bias; const float* xin; float* y;
  __device__ void operator()(int R, int col, float v) const {
    const int t  = R & 63, win = R >> 6;
    const int wx = win % kNWx, wy = (win / kNWx) % kNWx, bb = win / kNWimg;
    const int hh = (wy * kWS + (t >> 3) + kShift) % kHW;
    const int ww = (wx * kWS + (t & 7)  + kShift) % kHW;
    const size_t tok = (size_t)bb * kT + (size_t)hh * kHW + ww;
    y[tok * kC + col] = xin[tok * kC + col] + v + bias[col];
  }
};
struct EpiGelu {                // + fc1 bias, exact gelu, f16 out [M][384]
  const float* bias; _Float16* out;
  __device__ void operator()(int row, int col, float v) const {
    const float u = v + bias[col];
    const float ge = 0.5f * u * (1.f + erff(u * 0.70710678118654752f));
    out[(size_t)row * kHID + col] = (_Float16)ge;
  }
};
struct EpiOut {                 // + fc2 bias + residual y, f32 out
  const float* bias; const float* y; float* out;
  __device__ void operator()(int row, int col, float v) const {
    out[(size_t)row * kC + col] = y[(size_t)row * kC + col] + v + bias[col];
  }
};

// ---------------------------------------------------------------------------
// Windowed attention: one block (4 waves) per (window, head).
// q/k tiles (64x32, row pitch 576) DMA'd into LDS by the TDM; v staged
// transposed manually. S = q @ k^T via WMMA (K=32) with the 1/sqrt(dh) scale
// folded into the score epilogue, + rel-pos bias + shift mask, LDS softmax,
// then P @ v via WMMA.
// ---------------------------------------------------------------------------
__global__ __launch_bounds__(128) void k_attn(
    const _Float16* __restrict__ qkv, const float* __restrict__ biasTab,
    _Float16* __restrict__ xattn) {
  __shared__ __align__(16) _Float16 sq[kN * kDH];     // q tile (unscaled)
  __shared__ __align__(16) _Float16 sk[kN * kDH];     // k tile
  __shared__ __align__(16) _Float16 svT[kDH * kN];    // v transposed [d][t]
  __shared__ __align__(16) float    sS[kN * kN];      // scores (f32)
  __shared__ __align__(16) _Float16 sP[kN * kN];      // softmax probs (f16)
  __shared__ int cls[kN];                             // shift-mask region id

  const int w = blockIdx.x;
  const int h = blockIdx.y;
  const _Float16* base = qkv + (size_t)w * kN * kQKN;

  const int wave = threadIdx.x >> 5;
  const int lane = threadIdx.x & 31;

  // TDM: q and k head tiles (64 rows x 32 elems, pitch 576) straight to LDS.
  if (wave == 0) {
    tdm_load_2d_f16(sq, base + h * kDH,      kDH, kN, kQKN);
    tdm_load_2d_f16(sk, base + kC + h * kDH, kDH, kN, kQKN);
    __builtin_amdgcn_s_wait_tensorcnt(0);
  }
  // v needs a transpose for contiguous B fragments -> manual staging.
  for (int i = threadIdx.x; i < kN * 4; i += 128) {
    const int t = i >> 2, p = i & 3;
    v8h vv = *(const v8h*)(base + (size_t)t * kQKN + 2 * kC + h * kDH + p * 8);
#pragma unroll
    for (int e = 0; e < 8; e++) svT[(p * 8 + e) * kN + t] = vv[e];
  }
  // Shift-mask region class per token (only last window row/col are mixed).
  if (threadIdx.x < kN) {
    const int t  = threadIdx.x;
    const int wx = w % kNWx, wy = (w / kNWx) % kNWx;
    const int rr = wy * kWS + (t >> 3);
    const int cc = wx * kWS + (t & 7);
    const int cr = rr < kHW - kWS ? 0 : (rr < kHW - kShift ? 1 : 2);
    const int cw = cc < kHW - kWS ? 0 : (cc < kHW - kShift ? 1 : 2);
    cls[t] = cr * 3 + cw;
  }
  __syncthreads();

  const int mi = wave;                         // query 16-row strip per wave

  // S = (q @ k^T) * scale + bias + mask
  const v16h a = load_frag(sq + mi * 16 * kDH, kDH, lane);
  const float* bt = biasTab + (size_t)h * kN * kN;
  const int mb = mi * 16 + ((lane >> 4) << 3);
#pragma unroll
  for (int ni = 0; ni < 4; ni++) {
    v8f c = {};
    c = wmma_f16(a, load_frag(sk + ni * 16 * kDH, kDH, lane), c);
    const int ncol = ni * 16 + (lane & 15);
#pragma unroll
    for (int r = 0; r < 8; r++) {
      const int m = mb + r;
      const float mv = (cls[m] == cls[ncol]) ? 0.f : -100.f;
      sS[m * kN + ncol] = c[r] * kScale + bt[m * kN + ncol] + mv;
    }
  }
  __syncthreads();

  // Row softmax over keys (one thread per query row).
  if (threadIdx.x < kN) {
    const int m = threadIdx.x;
    float mx = -3.4e38f;
    for (int n = 0; n < kN; n++) mx = fmaxf(mx, sS[m * kN + n]);
    float sum = 0.f;
    for (int n = 0; n < kN; n++) {
      const float e = __expf(sS[m * kN + n] - mx);
      sS[m * kN + n] = e; sum += e;
    }
    const float rs = 1.f / sum;
    for (int n = 0; n < kN; n++) sP[m * kN + n] = (_Float16)(sS[m * kN + n] * rs);
  }
  __syncthreads();

  // out = P @ v  (K=64 -> two WMMA steps), store f16 at col h*32+d.
  const v16h p0 = load_frag(sP + mi * 16 * kN + 0,  kN, lane);
  const v16h p1 = load_frag(sP + mi * 16 * kN + 32, kN, lane);
#pragma unroll
  for (int ni = 0; ni < 2; ni++) {
    v8f c = {};
    c = wmma_f16(p0, load_frag(svT + ni * 16 * kN + 0,  kN, lane), c);
    c = wmma_f16(p1, load_frag(svT + ni * 16 * kN + 32, kN, lane), c);
    const int col   = h * kDH + ni * 16 + (lane & 15);
    const int rbase = w * kN + mi * 16 + ((lane >> 4) << 3);
#pragma unroll
    for (int r = 0; r < 8; r++)
      xattn[(size_t)(rbase + r) * kC + col] = (_Float16)c[r];
  }
}

// ---------------------------------------------------------------------------
extern "C" void kernel_launch(void* const* d_in, const int* in_sizes, int n_in,
                              void* d_out, int out_size, void* d_ws, size_t ws_size,
                              hipStream_t stream) {
  (void)in_sizes; (void)n_in; (void)out_size; (void)ws_size;
  const float* x      = (const float*)d_in[0];
  const float* n1g    = (const float*)d_in[1];
  const float* n1b    = (const float*)d_in[2];
  const float* qkv_w  = (const float*)d_in[3];
  const float* qkv_b  = (const float*)d_in[4];
  const float* proj_w = (const float*)d_in[5];
  const float* proj_b = (const float*)d_in[6];
  const float* relt   = (const float*)d_in[7];
  const float* n2g    = (const float*)d_in[8];
  const float* n2b    = (const float*)d_in[9];
  const float* fc1_w  = (const float*)d_in[10];
  const float* fc1_b  = (const float*)d_in[11];
  const float* fc2_w  = (const float*)d_in[12];
  const float* fc2_b  = (const float*)d_in[13];
  float* out = (float*)d_out;

  // Workspace layout (256B aligned). Aliasing: h16 reuses xw (dead after QKV
  // GEMM); fc1 activations reuse the QKV buffer (dead after attention).
  char* ws = (char*)d_ws;
  size_t o = 0;
  auto take = [&](size_t bytes) {
    size_t p = o; o += (bytes + 255) & ~(size_t)255; return p;
  };
  _Float16* qkvT  = (_Float16*)(ws + take((size_t)kQKN * kC * 2));
  _Float16* projT = (_Float16*)(ws + take((size_t)kC * kC * 2));
  _Float16* fc1T  = (_Float16*)(ws + take((size_t)kHID * kC * 2));
  _Float16* fc2T  = (_Float16*)(ws + take((size_t)kC * kHID * 2));
  float*    biasT = (float*)   (ws + take((size_t)kNH * kN * kN * 4));
  _Float16* xw    = (_Float16*)(ws + take((size_t)kM * kC * 2));    // -> h16
  _Float16* qkv16 = (_Float16*)(ws + take((size_t)kM * kQKN * 2));  // -> a16
  _Float16* xattn = (_Float16*)(ws + take((size_t)kM * kC * 2));
  float*    ybuf  = (float*)   (ws + take((size_t)kM * kC * 4));
  _Float16* h16 = xw;
  _Float16* a16 = qkv16;

  // Weight / bias-table prep (tiny).
  k_wprep<<<dim3((kC * kQKN + 255) / 256), 256, 0, stream>>>(qkv_w, qkvT, kC, kQKN);
  k_wprep<<<dim3((kC * kC   + 255) / 256), 256, 0, stream>>>(proj_w, projT, kC, kC);
  k_wprep<<<dim3((kC * kHID + 255) / 256), 256, 0, stream>>>(fc1_w, fc1T, kC, kHID);
  k_wprep<<<dim3((kHID * kC + 255) / 256), 256, 0, stream>>>(fc2_w, fc2T, kHID, kC);
  k_biasprep<<<dim3(kNH * kN * kN / 256), 256, 0, stream>>>(relt, biasT);

  // Main pipeline.
  k_ln1_window<<<dim3(kM / 8), 256, 0, stream>>>(x, n1g, n1b, xw);
  k_gemm64<kC, EpiQKV><<<dim3(kM / 64, kQKN / 64), 256, 0, stream>>>(
      xw, qkvT, EpiQKV{qkv_b, qkv16});
  k_attn<<<dim3(kNW, kNH), 128, 0, stream>>>(qkv16, biasT, xattn);
  k_gemm64<kC, EpiProj><<<dim3(kM / 64, kC / 64), 256, 0, stream>>>(
      xattn, projT, EpiProj{proj_b, x, ybuf});
  k_ln2<<<dim3(kM / 8), 256, 0, stream>>>(ybuf, n2g, n2b, h16);
  k_gemm64<kC, EpiGelu><<<dim3(kM / 64, kHID / 64), 256, 0, stream>>>(
      h16, fc1T, EpiGelu{fc1_b, a16});
  k_gemm64<kHID, EpiOut><<<dim3(kM / 64, kC / 64), 256, 0, stream>>>(
      a16, fc2T, EpiOut{fc2_b, ybuf, out});
}
